// TransformerEncoderBlock_18580028523068
// MI455X (gfx1250) — matrix operand order repaired and checked by reading, not compiled, not run on hardware
//
#include <hip/hip_runtime.h>
#include <math.h>

// ---------------------------------------------------------------------------
// Types / helpers
// ---------------------------------------------------------------------------
typedef __attribute__((ext_vector_type(16))) __bf16 v16bf;
typedef __attribute__((ext_vector_type(8)))  float  v8f;

#define DEV static __device__ __forceinline__

#if defined(__AMDGCN__) && __has_builtin(__builtin_amdgcn_global_load_async_to_lds_b128) && __has_builtin(__builtin_amdgcn_s_wait_asynccnt)
#define USE_ASYNC 1
typedef __attribute__((ext_vector_type(4))) int v4i_t;
typedef __attribute__((address_space(1))) v4i_t* as1_v4i_p;   // global int4*
typedef __attribute__((address_space(3))) v4i_t* as3_v4i_p;   // LDS    int4*
#else
#define USE_ASYNC 0
#endif

DEV unsigned short f2bf(float f) {
    unsigned int u = __float_as_uint(f);
    return (unsigned short)((u + 0x7FFFu + ((u >> 16) & 1u)) >> 16);  // RNE
}

struct U16x16 { uint4 lo; uint4 hi; };

DEV v16bf make_frag(uint4 lo, uint4 hi) {
    U16x16 t; t.lo = lo; t.hi = hi;
    return __builtin_bit_cast(v16bf, t);
}

DEV v8f wmma_bf16(v16bf a, v16bf b, v8f c) {
    // (neg_a, A, neg_b, B, c_mod, C, reuse_a, reuse_b)
    return __builtin_amdgcn_wmma_f32_16x16x32_bf16(
        false, a, false, b, (short)0, c, false, false);
}

// 16-byte global -> LDS copy: async (ASYNCcnt) on gfx1250, sync fallback.
DEV void cp_b128(unsigned short* lds, const unsigned short* g) {
#if USE_ASYNC
    __builtin_amdgcn_global_load_async_to_lds_b128(
        (as1_v4i_p)g, (as3_v4i_p)lds, 0, 0);
#else
    *(uint4*)lds = *(const uint4*)g;
#endif
}

DEV void cp_commit() {
#if USE_ASYNC
    __builtin_amdgcn_s_wait_asynccnt(0);
#endif
}

// ---------------------------------------------------------------------------
// LayerNorm (f32 in) -> bf16 out.  One 256-thread block per token row of 768.
// ---------------------------------------------------------------------------
__global__ __launch_bounds__(256) void ln_bf16_kernel(
    const float* __restrict__ x, const float* __restrict__ g,
    const float* __restrict__ b, unsigned short* __restrict__ out)
{
    const int row = blockIdx.x;
    const int tid = threadIdx.x;
    const float* xr = x + (size_t)row * 768;
    float v0 = xr[tid], v1 = xr[tid + 256], v2 = xr[tid + 512];
    __shared__ float r1[256], r2[256];
    r1[tid] = v0 + v1 + v2;
    r2[tid] = v0 * v0 + v1 * v1 + v2 * v2;
    __syncthreads();
    for (int o = 128; o > 0; o >>= 1) {
        if (tid < o) { r1[tid] += r1[tid + o]; r2[tid] += r2[tid + o]; }
        __syncthreads();
    }
    const float mu  = r1[0] * (1.0f / 768.0f);
    const float var = r2[0] * (1.0f / 768.0f) - mu * mu;
    const float rs  = rsqrtf(var + 1e-5f);
    unsigned short* orow = out + (size_t)row * 768;
    orow[tid]       = f2bf((v0 - mu) * rs * g[tid]       + b[tid]);
    orow[tid + 256] = f2bf((v1 - mu) * rs * g[tid + 256] + b[tid + 256]);
    orow[tid + 512] = f2bf((v2 - mu) * rs * g[tid + 512] + b[tid + 512]);
}

// ---------------------------------------------------------------------------
// Weight f32 [K x N]  ->  bf16 transposed [N x K]  (tiled via LDS)
// ---------------------------------------------------------------------------
__global__ __launch_bounds__(256) void transpose_bf16_kernel(
    const float* __restrict__ in, unsigned short* __restrict__ out,
    int K, int N)
{
    __shared__ float tile[32][33];
    const int tx = threadIdx.x, ty = threadIdx.y;     // (32, 8)
    const int n0 = blockIdx.x * 32, k0 = blockIdx.y * 32;
    #pragma unroll
    for (int j = 0; j < 32; j += 8)
        tile[ty + j][tx] = in[(size_t)(k0 + ty + j) * N + n0 + tx];
    __syncthreads();
    #pragma unroll
    for (int j = 0; j < 32; j += 8)
        out[(size_t)(n0 + ty + j) * K + k0 + tx] = f2bf(tile[tx][ty + j]);
}

// ---------------------------------------------------------------------------
// bf16 WMMA GEMM:  C[M,N] = A[M,K] * B[K,N]   (B given transposed: Bt[N,K])
// Block = 256 threads (8 waves), tile 128(M) x 64(N), BLK_K = 32,
// double-buffered LDS with async global->LDS staging.
// EPI: 0 = bf16 out; 1 = f32 out + bias + residual; 2 = bf16 out + bias + GELU
// ---------------------------------------------------------------------------
template <int EPI>
__global__ __launch_bounds__(256) void gemm_bf16_wmma(
    const unsigned short* __restrict__ A,
    const unsigned short* __restrict__ Bt,
    const float* __restrict__ bias,
    const float* __restrict__ resid,
    void* __restrict__ out,
    int M, int N, int K)
{
    constexpr int LDT = 40;                 // padded row stride (conflict-free)
    __shared__ unsigned short As[2][128 * LDT];
    __shared__ unsigned short Bs[2][64 * LDT];

    const int tid  = threadIdx.x;
    const int wave = tid >> 5;              // 0..7 -> 16-row slice each
    const int lane = tid & 31;
    const int half = lane >> 4;
    const int l16  = lane & 15;
    const int m0   = blockIdx.y * 128;
    const int n0   = blockIdx.x * 64;

    // Stage one 128x32 A tile + 64x32 B tile into buffer `buf`.
    auto stage = [&](int buf, int k0) {
        // A: 512 b128 chunks, 2 per thread.
        #pragma unroll
        for (int i = 0; i < 2; ++i) {
            const int c = tid + i * 256;
            const int r = c >> 2, q = c & 3;
            cp_b128(&As[buf][r * LDT + q * 8],
                    &A[(size_t)(m0 + r) * K + k0 + q * 8]);
        }
        // B: 256 b128 chunks, 1 per thread.
        {
            const int r = tid >> 2, q = tid & 3;
            cp_b128(&Bs[buf][r * LDT + q * 8],
                    &Bt[(size_t)(n0 + r) * K + k0 + q * 8]);
        }
    };

    v8f acc[4] = {};
    const int nt = K >> 5;

    stage(0, 0);
    cp_commit();
    __syncthreads();

    for (int kt = 0; kt < nt; ++kt) {
        const int cur = kt & 1;
        if (kt + 1 < nt) stage(cur ^ 1, (kt + 1) << 5);   // overlap with compute

        const int mloc = wave * 16 + l16;
        v16bf af = make_frag(*(const uint4*)&As[cur][mloc * LDT + half * 8],
                             *(const uint4*)&As[cur][mloc * LDT + half * 8 + 16]);
        #pragma unroll
        for (int ns = 0; ns < 4; ++ns) {
            const unsigned short* bp = &Bs[cur][(ns * 16 + l16) * LDT + half * 8];
            v16bf bf_ = make_frag(*(const uint4*)bp, *(const uint4*)(bp + 16));
            acc[ns] = wmma_bf16(af, bf_, acc[ns]);
        }

        if (kt + 1 < nt) {
            cp_commit();        // our async loads for the next buffer done
            __syncthreads();    // everyone's loads visible / old buffer free
        }
    }

    // Epilogue.  C layout: VGPR r -> row (r + 8*half), col = lane%16.
    const int mg = m0 + wave * 16 + half * 8;
    #pragma unroll
    for (int r = 0; r < 8; ++r) {
        const int row = mg + r;
        #pragma unroll
        for (int ns = 0; ns < 4; ++ns) {
            const int col = n0 + ns * 16 + l16;
            float v = acc[ns][r];
            if constexpr (EPI == 0) {
                ((unsigned short*)out)[(size_t)row * N + col] = f2bf(v);
            } else if constexpr (EPI == 1) {
                ((float*)out)[(size_t)row * N + col] =
                    v + bias[col] + resid[(size_t)row * N + col];
            } else {
                float t = v + bias[col];
                t = 0.5f * t * (1.0f + erff(t * 0.70710678118654752f));
                ((unsigned short*)out)[(size_t)row * N + col] = f2bf(t);
            }
        }
    }
}

// ---------------------------------------------------------------------------
// Flash-style attention.  qkv bf16 [B*1024, 2304] (q|k|v each 768 = 12h x 64d)
// One block (4 waves) per (b, h, 64-query tile); streams 64-key blocks.
// scores scaled by 1/sqrt(768) per the reference.
// ---------------------------------------------------------------------------
__global__ __launch_bounds__(128) void attn_kernel(
    const unsigned short* __restrict__ qkv,
    unsigned short* __restrict__ outp)
{
    constexpr int LDQ = 72;   // 64 + pad (halfwords)
    __shared__ unsigned short Qs[64 * LDQ];
    __shared__ unsigned short Ks[64 * LDQ];
    __shared__ unsigned short Vs[64 * LDQ];      // transposed: [d][key]
    __shared__ unsigned short Ps[4 * 16 * LDQ];  // per-wave P tiles [q][key]

    const int tid  = threadIdx.x;
    const int wave = tid >> 5;
    const int lane = tid & 31;
    const int half = lane >> 4;
    const int l16  = lane & 15;

    const int id = blockIdx.x;
    const int qb = id & 15;
    const int h  = (id >> 4) % 12;
    const int b  = id / (16 * 12);
    const int q0 = qb * 64;
    const size_t tok0 = (size_t)b * 1024;

    // Stage Q tile [64 q x 64 d] (async).
    for (int c = tid; c < 512; c += 128) {
        const int r = c >> 3, q = c & 7;
        cp_b128(&Qs[r * LDQ + q * 8],
                &qkv[(tok0 + q0 + r) * 2304 + h * 64 + q * 8]);
    }
    cp_commit();
    __syncthreads();

    const int mloc = wave * 16 + l16;
    v16bf aq[2];
    #pragma unroll
    for (int kk = 0; kk < 2; ++kk)
        aq[kk] = make_frag(
            *(const uint4*)&Qs[mloc * LDQ + kk * 32 + half * 8],
            *(const uint4*)&Qs[mloc * LDQ + kk * 32 + half * 8 + 16]);

    float mrun[8], lrun[8];
    #pragma unroll
    for (int r = 0; r < 8; ++r) { mrun[r] = -3.0e38f; lrun[r] = 0.0f; }
    v8f ofrag[4] = {};

    const float scale = 0.036084391824351615f;   // 1/sqrt(768)
    unsigned short* pbase = &Ps[wave * 16 * LDQ];

    for (int j0 = 0; j0 < 1024; j0 += 64) {
        __syncthreads();
        // Stage K block [64 key x 64 d] (async; rows are B-columns for Q*K^T).
        for (int c = tid; c < 512; c += 128) {
            const int r = c >> 3, q = c & 7;
            cp_b128(&Ks[r * LDQ + q * 8],
                    &qkv[(tok0 + j0 + r) * 2304 + 768 + h * 64 + q * 8]);
        }
        // Stage V block transposed: Vs[d][key] (scalar transpose).
        for (int c = tid; c < 2048; c += 128) {
            const int r = c & 63, dp = c >> 6;   // key, d/2
            const unsigned int w2 = *(const unsigned int*)
                &qkv[(tok0 + j0 + r) * 2304 + 1536 + h * 64 + dp * 2];
            Vs[(dp * 2) * LDQ + r]     = (unsigned short)(w2 & 0xFFFFu);
            Vs[(dp * 2 + 1) * LDQ + r] = (unsigned short)(w2 >> 16);
        }
        cp_commit();
        __syncthreads();

        // S = Q * K^T  (four 16-key column groups, two d k-steps each)
        v8f sv[4] = {};
        #pragma unroll
        for (int ns = 0; ns < 4; ++ns) {
            #pragma unroll
            for (int kk = 0; kk < 2; ++kk) {
                const unsigned short* bp =
                    &Ks[(ns * 16 + l16) * LDQ + kk * 32 + half * 8];
                sv[ns] = wmma_bf16(
                    aq[kk],
                    make_frag(*(const uint4*)bp, *(const uint4*)(bp + 16)),
                    sv[ns]);
            }
        }

        // Streaming softmax update (rows live across 16-lane groups).
        #pragma unroll
        for (int r = 0; r < 8; ++r) {
            float xv[4];
            #pragma unroll
            for (int ns = 0; ns < 4; ++ns) xv[ns] = sv[ns][r] * scale;
            float t = fmaxf(fmaxf(xv[0], xv[1]), fmaxf(xv[2], xv[3]));
            #pragma unroll
            for (int o = 1; o < 16; o <<= 1) t = fmaxf(t, __shfl_xor(t, o, 32));
            const float mnew  = fmaxf(mrun[r], t);
            const float alpha = __expf(mrun[r] - mnew);
            float rs = 0.0f;
            #pragma unroll
            for (int ns = 0; ns < 4; ++ns) {
                xv[ns] = __expf(xv[ns] - mnew);
                rs += xv[ns];
            }
            #pragma unroll
            for (int o = 1; o < 16; o <<= 1) rs += __shfl_xor(rs, o, 32);
            lrun[r] = lrun[r] * alpha + rs;
            mrun[r] = mnew;
            ofrag[0][r] *= alpha; ofrag[1][r] *= alpha;
            ofrag[2][r] *= alpha; ofrag[3][r] *= alpha;
            #pragma unroll
            for (int ns = 0; ns < 4; ++ns)
                pbase[(r + half * 8) * LDQ + ns * 16 + l16] = f2bf(xv[ns]);
        }
        __syncthreads();

        // O += P * V  (two 32-key k-steps)
        v16bf ap[2];
        #pragma unroll
        for (int kk = 0; kk < 2; ++kk)
            ap[kk] = make_frag(
                *(const uint4*)&pbase[l16 * LDQ + kk * 32 + half * 8],
                *(const uint4*)&pbase[l16 * LDQ + kk * 32 + half * 8 + 16]);
        #pragma unroll
        for (int ns = 0; ns < 4; ++ns) {
            #pragma unroll
            for (int kk = 0; kk < 2; ++kk) {
                const unsigned short* vp =
                    &Vs[(ns * 16 + l16) * LDQ + kk * 32 + half * 8];
                ofrag[ns] = wmma_bf16(
                    ap[kk],
                    make_frag(*(const uint4*)vp, *(const uint4*)(vp + 16)),
                    ofrag[ns]);
            }
        }
    }

    // Normalize and write out[(b,q), h*64 + d] as bf16.
    #pragma unroll
    for (int r = 0; r < 8; ++r) {
        const float inv = 1.0f / lrun[r];
        const int qrow = q0 + wave * 16 + half * 8 + r;
        unsigned short* orow = outp + (tok0 + qrow) * 768 + h * 64;
        #pragma unroll
        for (int ns = 0; ns < 4; ++ns)
            orow[ns * 16 + l16] = f2bf(ofrag[ns][r] * inv);
    }
}

// ---------------------------------------------------------------------------
// Launch
// ---------------------------------------------------------------------------
extern "C" void kernel_launch(void* const* d_in, const int* in_sizes, int n_in,
                              void* d_out, int out_size, void* d_ws, size_t ws_size,
                              hipStream_t stream) {
    (void)in_sizes; (void)n_in; (void)out_size; (void)ws_size;

    const float* x     = (const float*)d_in[0];
    const float* ln1g  = (const float*)d_in[1];
    const float* ln1b  = (const float*)d_in[2];
    const float* wqkv  = (const float*)d_in[3];
    const float* wproj = (const float*)d_in[4];
    const float* bproj = (const float*)d_in[5];
    const float* ln2g  = (const float*)d_in[6];
    const float* ln2b  = (const float*)d_in[7];
    const float* wfc1  = (const float*)d_in[8];
    const float* bfc1  = (const float*)d_in[9];
    const float* wfc2  = (const float*)d_in[10];
    const float* bfc2  = (const float*)d_in[11];
    float* out = (float*)d_out;

    const int T = 8192;   // B*N tokens

    char* ws = (char*)d_ws;
    auto alloc = [&](size_t bytes) {
        char* p = ws; ws += (bytes + 255) & ~(size_t)255; return p;
    };
    unsigned short* wqkvT = (unsigned short*)alloc((size_t)2304 * 768 * 2);
    unsigned short* wprojT= (unsigned short*)alloc((size_t)768  * 768 * 2);
    unsigned short* wfc1T = (unsigned short*)alloc((size_t)3072 * 768 * 2);
    unsigned short* wfc2T = (unsigned short*)alloc((size_t)768  * 3072 * 2);
    unsigned short* h1    = (unsigned short*)alloc((size_t)T * 768  * 2);
    unsigned short* qkvb  = (unsigned short*)alloc((size_t)T * 2304 * 2);
    unsigned short* attnb = (unsigned short*)alloc((size_t)T * 768  * 2);
    float*          x1    = (float*)         alloc((size_t)T * 768  * 4);
    unsigned short* h2    = (unsigned short*)alloc((size_t)T * 768  * 2);
    unsigned short* mlp   = (unsigned short*)alloc((size_t)T * 3072 * 2);

    const dim3 tb(32, 8);
    transpose_bf16_kernel<<<dim3(2304/32, 768/32),  tb, 0, stream>>>(wqkv,  wqkvT, 768, 2304);
    transpose_bf16_kernel<<<dim3(768/32,  768/32),  tb, 0, stream>>>(wproj, wprojT,768, 768);
    transpose_bf16_kernel<<<dim3(3072/32, 768/32),  tb, 0, stream>>>(wfc1,  wfc1T, 768, 3072);
    transpose_bf16_kernel<<<dim3(768/32,  3072/32), tb, 0, stream>>>(wfc2,  wfc2T, 3072, 768);

    // LN1 -> h1 (bf16)
    ln_bf16_kernel<<<T, 256, 0, stream>>>(x, ln1g, ln1b, h1);

    // qkv = h1 @ w_qkv   (no bias)
    gemm_bf16_wmma<0><<<dim3(2304/64, T/128), 256, 0, stream>>>(
        h1, wqkvT, nullptr, nullptr, qkvb, T, 2304, 768);

    // attention -> attnb (bf16, [token][h*64+d])
    attn_kernel<<<8 * 12 * 16, 128, 0, stream>>>(qkvb, attnb);

    // x1 = x + attnb @ w_proj + b_proj   (f32)
    gemm_bf16_wmma<1><<<dim3(768/64, T/128), 256, 0, stream>>>(
        attnb, wprojT, bproj, x, x1, T, 768, 768);

    // LN2 -> h2 (bf16)
    ln_bf16_kernel<<<T, 256, 0, stream>>>(x1, ln2g, ln2b, h2);

    // mlp = gelu(h2 @ w_fc1 + b_fc1)  (bf16)
    gemm_bf16_wmma<2><<<dim3(3072/64, T/128), 256, 0, stream>>>(
        h2, wfc1T, bfc1, nullptr, mlp, T, 3072, 768);

    // out = x1 + mlp @ w_fc2 + b_fc2  (f32)
    gemm_bf16_wmma<1><<<dim3(768/64, T/128), 256, 0, stream>>>(
        mlp, wfc2T, bfc2, x1, out, T, 768, 3072);
}
